// NeighborhoodAttentionBlock_31190052503667
// MI455X (gfx1250) — matrix operand order repaired and checked by reading, not compiled, not run on hardware
//
#include <hip/hip_runtime.h>
#include <hip/hip_bf16.h>
#include <math.h>

// ---------------------------------------------------------------------------
// NeighborhoodAttentionBlock for MI455X (gfx1250), wave32 + WMMA bf16.
//   x:(2,64,64,256) f32.  All GEMMs via v_wmma_f32_16x16x32_bf16 with
//   32x64 per-wave register tiling; B panel staged in LDS (ds_load_b128)
//   so only the prefetched A stream uses loadcnt.
// ---------------------------------------------------------------------------

typedef __bf16 bf16_t;
typedef __attribute__((ext_vector_type(8)))  bf16_t v8bf;
typedef __attribute__((ext_vector_type(16))) bf16_t v16bf;
typedef __attribute__((ext_vector_type(8)))  float  v8f;

#define TOKENS 8192      // B*H*W = 2*64*64
#define CDIM   256
#define NHEAD  4
#define HD     64
#define NNB    49        // 7*7, always full window due to clip
#define GK     256       // K dim of every GEMM in this block
#define GKCN   (GK / 32) // 8 K-steps

// ---------------- wave32 reductions ----------------
__device__ __forceinline__ float wave_sum(float v) {
  #pragma unroll
  for (int o = 16; o > 0; o >>= 1) v += __shfl_xor(v, o, 32);
  return v;
}
__device__ __forceinline__ float wave_max(float v) {
  #pragma unroll
  for (int o = 16; o > 0; o >>= 1) v = fmaxf(v, __shfl_xor(v, o, 32));
  return v;
}

// ---------------------------------------------------------------------------
// Weight packing: f32 (K x N, row-major) -> bf16 fragment-major for WMMA B.
// Layout: for (ntile, kc): 32 lanes * 16 contiguous bf16.
//   lane L (col n = ntile*16 + (L&15)) holds K = kc*32 + 16*(L>>4) + t, t=0..15
// ---------------------------------------------------------------------------
__global__ void pack_weight(const float* __restrict__ w, bf16_t* __restrict__ p,
                            int K, int N) {
  int tid = blockIdx.x * 256 + threadIdx.x;
  if (tid >= K * N) return;
  int t     = tid & 15;
  int lane  = (tid >> 4) & 31;
  int rest  = tid >> 9;
  int kcn   = K / 32;
  int kc    = rest % kcn;
  int ntile = rest / kcn;
  int n = ntile * 16 + (lane & 15);
  int k = kc * 32 + 16 * (lane >> 4) + t;
  p[tid] = (bf16_t)w[(size_t)k * N + n];
}

// ---------------------------------------------------------------------------
// Fused RMSNorm (+ optional RoPE2D) -> bf16.  One wave per token, 8 ch/lane.
// ---------------------------------------------------------------------------
template <bool ROPE>
__global__ void rmsnorm_kernel(const float* __restrict__ x,
                               const float* __restrict__ w,
                               bf16_t* __restrict__ out) {
  int lane = threadIdx.x & 31;
  int wave = threadIdx.x >> 5;
  int tok  = blockIdx.x * 8 + wave;
  const float* px = x + (size_t)tok * CDIM + lane * 8;
  float v[8];
  #pragma unroll
  for (int i = 0; i < 8; ++i) v[i] = px[i];
  float ss = 0.f;
  #pragma unroll
  for (int i = 0; i < 8; ++i) ss += v[i] * v[i];
  ss = wave_sum(ss);
  float r = rsqrtf(ss * (1.f / 256.f) + 1e-6f);
  const float* pw = w + lane * 8;
  float n[8], o[8];
  #pragma unroll
  for (int i = 0; i < 8; ++i) n[i] = v[i] * r * pw[i];
  if (ROPE) {
    int xq = tok & 63, yq = (tok >> 6) & 63;
    int ch0 = lane * 8;
    #pragma unroll
    for (int p = 0; p < 4; ++p) {
      int c0 = ch0 + 2 * p;
      int j = (c0 >> 1) & 63;                       // pair index within half
      float pos = (c0 < 128) ? (float)yq : (float)xq;
      float th = pos * __expf(-(float)j * 0.14391157f);  // 10000^(-j/64)
      float sn, cs;
      __sincosf(th, &sn, &cs);
      o[2 * p]     = n[2 * p] * cs - n[2 * p + 1] * sn;
      o[2 * p + 1] = n[2 * p] * sn + n[2 * p + 1] * cs;
    }
  } else {
    #pragma unroll
    for (int i = 0; i < 8; ++i) o[i] = n[i];
  }
  bf16_t* po = out + (size_t)tok * CDIM + lane * 8;
  v8bf ob;
  #pragma unroll
  for (int i = 0; i < 8; ++i) ob[i] = (bf16_t)o[i];
  *(v8bf*)po = ob;
}

// ---------------------------------------------------------------------------
// WMMA GEMM: out[M,N] = A[M,K=256](bf16) * B(packed bf16) (+bias)(+residual)
// Wave tile = 32 rows x 64 cols: 2 A-fragments x 4 B-fragments -> 8
// independent accumulators, 8 wmma per K-step.  Block = 8 waves = 256x64.
// B panel (4 ntiles x 8 kc x 1KB = 32KB, contiguous in packed layout) is
// staged in LDS once; inner loop reads B via ds_load_b128.
// A layout (16-bit A 16x32): lane L row = m0+(L&15);
//   VGPR0-7 = K kb..kb+15, kb = kc*32 + 16*(L>>4).
// MODE: 0 = plain store, 1 = +bias, 2 = +bias+residual
// ---------------------------------------------------------------------------
template <int MODE>
__global__ void gemm_wmma_bf16(const bf16_t* __restrict__ A,
                               const bf16_t* __restrict__ Bp,
                               const float* __restrict__ bias,
                               const float* __restrict__ resid,
                               float* __restrict__ out,
                               int N) {
  __shared__ __align__(16) bf16_t Bs[GKCN * 4 * 512];   // 32 KB
  int lane = threadIdx.x & 31;
  int wave = threadIdx.x >> 5;
  int m0 = blockIdx.y * 256 + wave * 32;
  int kg = lane >> 4;

  // ---- cooperative stage of the 32KB contiguous B panel into LDS ----
  {
    const bf16_t* bsrc = Bp + (size_t)blockIdx.x * 4 * GKCN * 512;
    #pragma unroll
    for (int i = 0; i < 8; ++i) {
      int e = (threadIdx.x + i * 256) * 8;            // v8bf chunk
      *(v8bf*)&Bs[e] = *(const v8bf*)(bsrc + e);
    }
  }
  __syncthreads();

  const bf16_t* aptr = A + (size_t)(m0 + (lane & 15)) * GK + kg * 16;
  const bf16_t* bls = &Bs[lane * 16];
  const size_t arow16 = (size_t)16 * GK;              // 16 rows down

  v8f acc[2][4];
  #pragma unroll
  for (int mi = 0; mi < 2; ++mi)
    #pragma unroll
    for (int nt = 0; nt < 4; ++nt)
      acc[mi][nt] = v8f{0.f, 0.f, 0.f, 0.f, 0.f, 0.f, 0.f, 0.f};

  for (int kc = 0; kc < GKCN; ++kc) {
    // prefetch A two K-steps ahead (global_prefetch_b8, WGP scope)
    __builtin_prefetch(aptr + 64, 0, 3);
    __builtin_prefetch(aptr + arow16 + 64, 0, 3);

    v8bf a0lo = *(const v8bf*)(aptr);
    v8bf a0hi = *(const v8bf*)(aptr + 8);
    v8bf a1lo = *(const v8bf*)(aptr + arow16);
    v8bf a1hi = *(const v8bf*)(aptr + arow16 + 8);
    v16bf av0 = __builtin_shufflevector(a0lo, a0hi, 0, 1, 2, 3, 4, 5, 6, 7,
                                        8, 9, 10, 11, 12, 13, 14, 15);
    v16bf av1 = __builtin_shufflevector(a1lo, a1hi, 0, 1, 2, 3, 4, 5, 6, 7,
                                        8, 9, 10, 11, 12, 13, 14, 15);
    #pragma unroll
    for (int nt = 0; nt < 4; ++nt) {
      const bf16_t* bp = bls + (nt * GKCN + kc) * 512;    // ds_load_b128 x2
      v8bf blo = *(const v8bf*)(bp);
      v8bf bhi = *(const v8bf*)(bp + 8);
      v16bf bv = __builtin_shufflevector(blo, bhi, 0, 1, 2, 3, 4, 5, 6, 7,
                                         8, 9, 10, 11, 12, 13, 14, 15);
      acc[0][nt] = __builtin_amdgcn_wmma_f32_16x16x32_bf16(
          false, av0, false, bv, (short)0, acc[0][nt], false, false);
      acc[1][nt] = __builtin_amdgcn_wmma_f32_16x16x32_bf16(
          false, av1, false, bv, (short)0, acc[1][nt], false, false);
    }
    aptr += 32;
  }

  int colb = blockIdx.x * 64 + (lane & 15);
  #pragma unroll
  for (int mi = 0; mi < 2; ++mi) {
    int rbase = m0 + mi * 16 + 8 * kg;
    #pragma unroll
    for (int nt = 0; nt < 4; ++nt) {
      int col = colb + nt * 16;
      float bv0 = (MODE >= 1) ? bias[col] : 0.f;
      #pragma unroll
      for (int v = 0; v < 8; ++v) {
        int row = rbase + v;
        float val = acc[mi][nt][v] + bv0;
        if (MODE == 2) val += resid[(size_t)row * N + col];
        out[(size_t)row * N + col] = val;
      }
    }
  }
}

// ---------------------------------------------------------------------------
// 7x7 neighborhood attention.  qkv: [tok, 768] f32 (q|k|v, head-major 64).
// One wave per (token, head); 8 waves per block.
// ---------------------------------------------------------------------------
__global__ void neigh_attn(const float* __restrict__ qkv,
                           bf16_t* __restrict__ attn_out) {
  __shared__ __align__(16) float qs[8][64];
  __shared__ float wts[8][64];
  int lane = threadIdx.x & 31;
  int wave = threadIdx.x >> 5;
  int gid  = blockIdx.x * 8 + wave;
  int head = gid & 3;
  int tok  = gid >> 2;                 // b*4096 + y*64 + x
  int xq = tok & 63, yq = (tok >> 6) & 63;
  int tokbase = tok & ~4095;           // b*4096
  int wy0 = min(max(yq - 3, 0), 57);
  int wx0 = min(max(xq - 3, 0), 57);

  // stage q (pre-scaled by 1/sqrt(64)) into LDS
  const float* qp = qkv + (size_t)tok * 768 + head * HD;
  qs[wave][lane]      = qp[lane] * 0.125f;
  qs[wave][lane + 32] = qp[lane + 32] * 0.125f;
  __syncthreads();

  // logits: lanes over neighbors (lane and lane+32)
  float l0 = -1e30f, l1 = -1e30f;
  {
    int nb = lane;
    int ky = wy0 + nb / 7, kx = wx0 + nb % 7;
    const float* kp = qkv + (size_t)(tokbase + ky * 64 + kx) * 768 + 256 + head * HD;
    float d = 0.f;
    #pragma unroll
    for (int i = 0; i < 64; i += 4) {
      float4 kk = *(const float4*)(kp + i);
      float4 qq = *(const float4*)(&qs[wave][i]);
      d += kk.x * qq.x + kk.y * qq.y + kk.z * qq.z + kk.w * qq.w;
    }
    l0 = d;
  }
  if (lane < NNB - 32) {
    int nb = lane + 32;
    int ky = wy0 + nb / 7, kx = wx0 + nb % 7;
    const float* kp = qkv + (size_t)(tokbase + ky * 64 + kx) * 768 + 256 + head * HD;
    float d = 0.f;
    #pragma unroll
    for (int i = 0; i < 64; i += 4) {
      float4 kk = *(const float4*)(kp + i);
      float4 qq = *(const float4*)(&qs[wave][i]);
      d += kk.x * qq.x + kk.y * qq.y + kk.z * qq.z + kk.w * qq.w;
    }
    l1 = d;
  }

  // wave32 softmax over 49 logits
  float mx = wave_max(fmaxf(l0, l1));
  float e0 = __expf(l0 - mx);
  float e1 = (lane < NNB - 32) ? __expf(l1 - mx) : 0.f;
  float s = wave_sum(e0 + e1);
  float inv = 1.f / s;
  wts[wave][lane] = e0 * inv;
  if (lane < NNB - 32) wts[wave][lane + 32] = e1 * inv;
  __syncthreads();

  // weighted V sum: lanes over head dims (lane, lane+32)
  float a0 = 0.f, a1 = 0.f;
  for (int nb = 0; nb < NNB; ++nb) {
    int ky = wy0 + nb / 7, kx = wx0 + nb % 7;
    const float* vp = qkv + (size_t)(tokbase + ky * 64 + kx) * 768 + 512 + head * HD;
    float wgt = wts[wave][nb];
    a0 += wgt * vp[lane];
    a1 += wgt * vp[lane + 32];
  }
  bf16_t* op = attn_out + (size_t)tok * CDIM + head * HD;
  op[lane]      = (bf16_t)a0;
  op[lane + 32] = (bf16_t)a1;
}

// ---------------------------------------------------------------------------
// GEGLU: g = gelu_exact(h[:, :256]) * h[:, 256:512] -> bf16
// ---------------------------------------------------------------------------
__global__ void geglu_kernel(const float* __restrict__ h, bf16_t* __restrict__ g) {
  int idx = blockIdx.x * 256 + threadIdx.x;     // TOKENS*256 total
  int tok = idx >> 8;
  int j = idx & 255;
  float gate = h[(size_t)tok * 512 + j];
  float val  = h[(size_t)tok * 512 + 256 + j];
  float ge = 0.5f * gate * (1.f + erff(gate * 0.70710678f));
  g[idx] = (bf16_t)(ge * val);
}

// ---------------------------------------------------------------------------
// Host launcher
// ---------------------------------------------------------------------------
extern "C" void kernel_launch(void* const* d_in, const int* in_sizes, int n_in,
                              void* d_out, int out_size, void* d_ws, size_t ws_size,
                              hipStream_t stream) {
  const float* x       = (const float*)d_in[0];
  const float* norm1_w = (const float*)d_in[1];
  const float* norm2_w = (const float*)d_in[2];
  const float* w_qkv   = (const float*)d_in[3];
  const float* w_proj  = (const float*)d_in[4];
  const float* b_proj  = (const float*)d_in[5];
  const float* ff1_w   = (const float*)d_in[6];
  const float* ff1_b   = (const float*)d_in[7];
  const float* ff2_w   = (const float*)d_in[8];
  const float* ff2_b   = (const float*)d_in[9];
  float* out = (float*)d_out;

  char* ws = (char*)d_ws;
  size_t off = 0;
  auto alloc = [&](size_t bytes) -> void* {
    void* p = ws + off;
    off += (bytes + 255) & ~(size_t)255;
    return p;
  };
  bf16_t* h_bf    = (bf16_t*)alloc((size_t)TOKENS * CDIM * 2);      // 4 MB
  bf16_t* pw_qkv  = (bf16_t*)alloc((size_t)CDIM * 768 * 2);
  bf16_t* pw_proj = (bf16_t*)alloc((size_t)CDIM * 256 * 2);
  bf16_t* pw_ff1  = (bf16_t*)alloc((size_t)CDIM * 512 * 2);
  bf16_t* pw_ff2  = (bf16_t*)alloc((size_t)CDIM * 256 * 2);
  float*  qkv     = (float*)alloc((size_t)TOKENS * 768 * 4);        // 24 MB
  bf16_t* attn_bf = (bf16_t*)alloc((size_t)TOKENS * CDIM * 2);      // 4 MB
  float*  x1      = (float*)alloc((size_t)TOKENS * CDIM * 4);       // 8 MB
  bf16_t* n2_bf   = (bf16_t*)alloc((size_t)TOKENS * CDIM * 2);      // 4 MB
  float*  hff     = (float*)alloc((size_t)TOKENS * 512 * 4);        // 16 MB
  bf16_t* g_bf    = (bf16_t*)alloc((size_t)TOKENS * CDIM * 2);      // 4 MB
  (void)ws_size;

  // 1) pack weights to bf16 WMMA-fragment layout
  hipLaunchKernelGGL(pack_weight, dim3(256 * 768 / 256), dim3(256), 0, stream, w_qkv, pw_qkv, 256, 768);
  hipLaunchKernelGGL(pack_weight, dim3(256 * 256 / 256), dim3(256), 0, stream, w_proj, pw_proj, 256, 256);
  hipLaunchKernelGGL(pack_weight, dim3(256 * 512 / 256), dim3(256), 0, stream, ff1_w, pw_ff1, 256, 512);
  hipLaunchKernelGGL(pack_weight, dim3(256 * 256 / 256), dim3(256), 0, stream, ff2_w, pw_ff2, 256, 256);

  // 2) h = rope2d(rmsnorm(x, norm1_w)) -> bf16
  hipLaunchKernelGGL((rmsnorm_kernel<true>), dim3(TOKENS / 8), dim3(256), 0, stream, x, norm1_w, h_bf);

  // 3) qkv = h @ w_qkv   (M=8192, K=256, N=768)
  hipLaunchKernelGGL((gemm_wmma_bf16<0>), dim3(768 / 64, TOKENS / 256), dim3(256), 0, stream,
                     h_bf, pw_qkv, (const float*)nullptr, (const float*)nullptr, qkv, 768);

  // 4) neighborhood attention -> bf16
  hipLaunchKernelGGL(neigh_attn, dim3(TOKENS * NHEAD / 8), dim3(256), 0, stream, qkv, attn_bf);

  // 5) x1 = attn @ w_proj + b_proj + x
  hipLaunchKernelGGL((gemm_wmma_bf16<2>), dim3(256 / 64, TOKENS / 256), dim3(256), 0, stream,
                     attn_bf, pw_proj, b_proj, x, x1, 256);

  // 6) n2 = rmsnorm(x1, norm2_w) -> bf16
  hipLaunchKernelGGL((rmsnorm_kernel<false>), dim3(TOKENS / 8), dim3(256), 0, stream, x1, norm2_w, n2_bf);

  // 7) hff = n2 @ ff1_w + ff1_b   (N=512)
  hipLaunchKernelGGL((gemm_wmma_bf16<1>), dim3(512 / 64, TOKENS / 256), dim3(256), 0, stream,
                     n2_bf, pw_ff1, ff1_b, (const float*)nullptr, hff, 512);

  // 8) g = gelu(gate) * value -> bf16
  hipLaunchKernelGGL(geglu_kernel, dim3(TOKENS * 256 / 256), dim3(256), 0, stream, hff, g_bf);

  // 9) out = g @ ff2_w + ff2_b + x1
  hipLaunchKernelGGL((gemm_wmma_bf16<2>), dim3(256 / 64, TOKENS / 256), dim3(256), 0, stream,
                     g_bf, pw_ff2, ff2_b, x1, out, 256);
}
